// RNN_s_7181185319527
// MI455X (gfx1250) — compile-verified, matrix operand order checked by reading, not run on hardware
//
#include <hip/hip_runtime.h>
#include <hip/hip_bf16.h>
#include <stdint.h>

// Dimensions (fixed by the reference)
#define B_ 64
#define S_ 1024
#define L_ 1024
#define N_ 256
#define O_ 11

typedef __attribute__((ext_vector_type(8)))  float        v8f;
typedef __attribute__((ext_vector_type(16))) __bf16       v16bf;
typedef __attribute__((ext_vector_type(4)))  unsigned int v4u;

// Hardware f32 -> bf16 conversion (lowers to v_cvt_pk_bf16_f32 / v_cvt_bf16)
__device__ __forceinline__ unsigned short f2bf(float f) {
  union { __bf16 h; unsigned short u; } c;
  c.h = (__bf16)f;
  return c.u;
}
__device__ __forceinline__ float bf2f(unsigned short h) {
  return __uint_as_float(((unsigned int)h) << 16);
}

// bf16 fragment: 16 bf16 per lane in K, two runs of 8 at p and p+16 (ISA A/B layout)
__device__ __forceinline__ v16bf load_b16(const unsigned short* p) {
  union { v16bf v; v4u q[2]; } f;
  f.q[0] = *(const v4u*)(p);
  f.q[1] = *(const v4u*)(p + 16);
  return f.v;
}
// A-tile with on-the-fly f32 -> bf16 hardware conversion (phase-1 GEMM)
__device__ __forceinline__ v16bf load_a_f32(const float* p) {
  union { v16bf v; __bf16 e[16]; } f;
#pragma unroll
  for (int i = 0; i < 8; ++i) f.e[i]     = (__bf16)p[i];
#pragma unroll
  for (int i = 0; i < 8; ++i) f.e[8 + i] = (__bf16)p[16 + i];
  return f.v;
}

// ---------------------------------------------------------------------------
// Kernel 1: convert W_i (256x1024 f32) to bf16 in workspace (L2-resident).
// ---------------------------------------------------------------------------
__global__ __launch_bounds__(256) void k_conv_wi(const float* __restrict__ wi,
                                                 unsigned short* __restrict__ wib) {
  int i = blockIdx.x * 256 + threadIdx.x;
  if (i < N_ * L_) wib[i] = f2bf(wi[i]);
}

// ---------------------------------------------------------------------------
// Kernel 2: D1[t*64+b][n] = input[b][t][:] @ W_i.T + b_i  for ALL timesteps.
// 65536x256x1024 bf16 WMMA GEMM. Each wave computes a 16x64 strip: one
// A-fragment (converted once) feeds 4 WMMAs -> 4x less A traffic & cvt VALU.
// Memory-bound by design: 256 MB input read once at 23.3 TB/s.
// ---------------------------------------------------------------------------
__global__ __launch_bounds__(256) void k_gemm_d1(
    const float* __restrict__ inp,            // [B][S][L] f32
    const unsigned short* __restrict__ wib,   // [N][L] bf16
    const float* __restrict__ bi,             // [N]
    float* __restrict__ d1)                   // [S*B][N] f32
{
  const int wave = threadIdx.x >> 5, lane = threadIdx.x & 31;
  const int r = lane & 15, half = lane >> 4;
  const int strip = blockIdx.x * 8 + wave;    // 0..16383 (4096 M-tiles x 4)
  const int mt = strip >> 2;                  // M-tile (16 rows)
  const int nq = strip & 3;                   // which 64-col quarter

  const int g  = mt * 16 + r;                 // A-fragment row for this lane
  const int t  = g >> 6, bb = g & 63;
  const float* arow = inp + ((size_t)bb * S_ + t) * L_;
  const unsigned short* brow0 = wib + (size_t)(nq * 64 +  0 + r) * L_;
  const unsigned short* brow1 = wib + (size_t)(nq * 64 + 16 + r) * L_;
  const unsigned short* brow2 = wib + (size_t)(nq * 64 + 32 + r) * L_;
  const unsigned short* brow3 = wib + (size_t)(nq * 64 + 48 + r) * L_;

  v8f acc0 = {0.f,0.f,0.f,0.f,0.f,0.f,0.f,0.f};
  v8f acc1 = acc0, acc2 = acc0, acc3 = acc0;
  for (int k0 = 0; k0 < L_; k0 += 32) {
    const int ka = k0 + half * 8;
    v16bf a = load_a_f32(arow + ka);
    acc0 = __builtin_amdgcn_wmma_f32_16x16x32_bf16(false, a, false, load_b16(brow0 + ka),
                                                   (short)0, acc0, false, false);
    acc1 = __builtin_amdgcn_wmma_f32_16x16x32_bf16(false, a, false, load_b16(brow1 + ka),
                                                   (short)0, acc1, false, false);
    acc2 = __builtin_amdgcn_wmma_f32_16x16x32_bf16(false, a, false, load_b16(brow2 + ka),
                                                   (short)0, acc2, false, false);
    acc3 = __builtin_amdgcn_wmma_f32_16x16x32_bf16(false, a, false, load_b16(brow3 + ka),
                                                   (short)0, acc3, false, false);
  }
  const int row0 = mt * 16 + half * 8;        // C/D layout: M = 8*half + j
  float* dout = d1 + (size_t)row0 * N_ + nq * 64 + r;
  const float b0 = bi[nq * 64 +  0 + r];
  const float b1 = bi[nq * 64 + 16 + r];
  const float b2 = bi[nq * 64 + 32 + r];
  const float b3 = bi[nq * 64 + 48 + r];
#pragma unroll
  for (int j = 0; j < 8; ++j) {
    dout[(size_t)j * N_ +  0] = acc0[j] + b0;
    dout[(size_t)j * N_ + 16] = acc1[j] + b1;
    dout[(size_t)j * N_ + 32] = acc2[j] + b2;
    dout[(size_t)j * N_ + 48] = acc3[j] + b3;
  }
}

// ---------------------------------------------------------------------------
// Kernel 3: sequential scan. One persistent 1024-thread workgroup (32 waves).
// LDS (exactly 320 KB, the CDNA5 cap):
//   [0,128K)   W_i2r bf16      [128K,256K) W_r bf16
//   [256K,288K) sp1 bf16       [288K,320K) sp2 bf16
// Softmax/reduction scratch ALIASES the sp1 region: sp1 lives (sync0..sync2),
// scratch lives (sync3..next sync0); layer-1 rewrites all of sp1 each step.
// ---------------------------------------------------------------------------
__global__ __launch_bounds__(1024) void k_scan(
    const float* __restrict__ d1,         // [S][B][N] precomputed
    const float* __restrict__ Wi2r, const float* __restrict__ bi2r,
    const float* __restrict__ Wr,   const float* __restrict__ br_b,
    const float* __restrict__ Wo,   const float* __restrict__ bo,
    const float* __restrict__ tau_adp_i, const float* __restrict__ tau_adp_r,
    const float* __restrict__ tau_m_i,   const float* __restrict__ tau_m_r,
    const float* __restrict__ tau_m_o,
    const float* __restrict__ init1, const float* __restrict__ init2,
    const float* __restrict__ init3,
    float* __restrict__ out)              // [S][B][O] log-probs + [1] fr
{
  extern __shared__ char smem[];
  unsigned short* wi2r_l = (unsigned short*)(smem);            // 128 KB
  unsigned short* wr_l   = (unsigned short*)(smem + 131072);   // 128 KB
  unsigned short* sp1l   = (unsigned short*)(smem + 262144);   //  32 KB (bf16)
  unsigned short* sp2l   = (unsigned short*)(smem + 294912);   //  32 KB (bf16)
  float*          mem3b  = (float*)(smem + 262144);            // aliases sp1
  float*          redb   = (float*)(smem + 262144 + 2816);     // aliases sp1

  const int tid  = threadIdx.x;
  const int wave = tid >> 5;
  const int lane = tid & 31;
  const int r    = lane & 15;
  const int half = lane >> 4;

  // Stage recurrent weights -> LDS as bf16 (once)
  for (int i = tid; i < N_ * N_; i += 1024) {
    wi2r_l[i] = f2bf(Wi2r[i]);
    wr_l[i]   = f2bf(Wr[i]);
  }
  // t=0 sp2 carry is the real-valued init2 tensor
  for (int i = tid; i < B_ * N_; i += 1024) sp2l[i] = f2bf(init2[i]);

  // ---- layer-1 state: 16 elements per thread, e = k*1024 + tid, n = tid&255
  const int n1 = tid & 255;
  const float alpha1 = __expf(-1.0f / tau_m_i[n1]);
  const float ro1    = __expf(-1.0f / tau_adp_i[n1]);
  float mem1[16], sp1f[16], b1[16];
#pragma unroll
  for (int k = 0; k < 16; ++k) {
    int e = k * 1024 + tid;
    mem1[k] = init1[e]; sp1f[k] = init1[e]; b1[k] = 0.1f;
  }

  // ---- layer-2 state: 2 tiles/wave x 8 rows/lane; fixed lane<->element map
  int   m0[2], ncol[2], rowb[2];
  float alpha2[2], ro2[2], bias2[2];
  float mem2[2][8], sp2f[2][8], b2[2][8];
#pragma unroll
  for (int i = 0; i < 2; ++i) {
    int tileid = wave * 2 + i;
    m0[i]   = (tileid >> 4) * 16;
    ncol[i] = (tileid & 15) * 16 + r;
    rowb[i] = m0[i] + half * 8;
    alpha2[i] = __expf(-1.0f / tau_m_r[ncol[i]]);
    ro2[i]    = __expf(-1.0f / tau_adp_r[ncol[i]]);
    bias2[i]  = bi2r[ncol[i]] + br_b[ncol[i]];
#pragma unroll
    for (int j = 0; j < 8; ++j) {
      float v = init2[(rowb[i] + j) * N_ + ncol[i]];
      mem2[i][j] = v; sp2f[i][j] = v; b2[i][j] = 0.1f;
    }
  }

  // ---- layer-3 state (64*11 = 704 readout neurons on threads 0..703)
  int b3 = 0, j3 = 0; float alpha3 = 0.f, mem3 = 0.f, bo3 = 0.f;
  if (tid < B_ * O_) {
    b3 = tid / O_; j3 = tid - b3 * O_;
    alpha3 = __expf(-1.0f / tau_m_o[j3]);
    mem3 = init3[tid]; bo3 = bo[j3];
  }

  float frc = 0.0f;

  for (int t = 0; t < S_; ++t) {
    __syncthreads();                               // sync0: prev-step LDS consumers done
    const float* d1t = d1 + (size_t)t * (B_ * N_);
    // ---- layer 1 (elementwise, uses precomputed d1); writes bf16 spikes
#pragma unroll
    for (int k = 0; k < 16; ++k) {
      int e = k * 1024 + tid;
      float inp = d1t[e];
      if (t + 1 < S_) __builtin_prefetch(d1t + (B_ * N_) + e, 0, 0);
      b1[k] = ro1 * b1[k] + (1.0f - ro1) * sp1f[k];
      float Bth = 0.1f + 1.84f * b1[k];
      mem1[k] = mem1[k] * alpha1 + (1.0f - alpha1) * inp - Bth * sp1f[k];
      float s = (mem1[k] - Bth) > 0.0f ? 1.0f : 0.0f;
      sp1f[k] = s; frc += s;
      sp1l[e] = s > 0.5f ? (unsigned short)0x3F80 : (unsigned short)0;
    }
    __syncthreads();                               // sync1: sp1 visible to all waves

    // ---- recurrent GEMM: r_in = sp1@Wi2r.T + sp2@Wr.T  (fused accumulate)
#pragma unroll
    for (int i = 0; i < 2; ++i) {
      v8f acc = {0.f,0.f,0.f,0.f,0.f,0.f,0.f,0.f};
      const unsigned short* ap1 = sp1l + (m0[i] + r) * N_;
      const unsigned short* ap2 = sp2l + (m0[i] + r) * N_;
      const unsigned short* bp1 = wi2r_l + ncol[i] * N_;
      const unsigned short* bp2 = wr_l   + ncol[i] * N_;
      for (int k0 = 0; k0 < N_; k0 += 32) {
        const int ka = k0 + half * 8;
        acc = __builtin_amdgcn_wmma_f32_16x16x32_bf16(false, load_b16(ap1 + ka),
                                                      false, load_b16(bp1 + ka),
                                                      (short)0, acc, false, false);
        acc = __builtin_amdgcn_wmma_f32_16x16x32_bf16(false, load_b16(ap2 + ka),
                                                      false, load_b16(bp2 + ka),
                                                      (short)0, acc, false, false);
      }
      // ---- layer 2 elementwise on this wave's fragment
#pragma unroll
      for (int j = 0; j < 8; ++j) {
        float rin = acc[j] + bias2[i];
        b2[i][j] = ro2[i] * b2[i][j] + (1.0f - ro2[i]) * sp2f[i][j];
        float Bth = 0.1f + 1.84f * b2[i][j];
        mem2[i][j] = mem2[i][j] * alpha2[i] + (1.0f - alpha2[i]) * rin - Bth * sp2f[i][j];
        float s = (mem2[i][j] - Bth) > 0.0f ? 1.0f : 0.0f;
        sp2f[i][j] = s; frc += s;
      }
    }
    __syncthreads();                               // sync2: all GEMM LDS reads done
#pragma unroll
    for (int i = 0; i < 2; ++i)
#pragma unroll
      for (int j = 0; j < 8; ++j)
        sp2l[(rowb[i] + j) * N_ + ncol[i]] =
            sp2f[i][j] > 0.5f ? (unsigned short)0x3F80 : (unsigned short)0;
    __syncthreads();                               // sync3: new sp2 visible

    // ---- layer 3 readout + leaky integrate
    if (tid < B_ * O_) {
      const float* wrow = Wo + j3 * N_;
      const unsigned short* srow = sp2l + b3 * N_;
      float o = 0.0f;
#pragma unroll 8
      for (int k = 0; k < N_; ++k) o += bf2f(srow[k]) * wrow[k];
      o += bo3;
      mem3 = mem3 * alpha3 + (1.0f - alpha3) * o;
      mem3b[tid] = mem3;
    }
    __syncthreads();                               // sync4
    // ---- log-softmax over the 11 outputs of each batch row
    if (tid < B_ * O_) {
      const float* rowp = mem3b + b3 * O_;
      float mx = rowp[0];
#pragma unroll
      for (int q = 1; q < O_; ++q) mx = fmaxf(mx, rowp[q]);
      float sum = 0.0f;
#pragma unroll
      for (int q = 0; q < O_; ++q) sum += __expf(rowp[q] - mx);
      out[(size_t)t * (B_ * O_) + tid] = mem3 - mx - __logf(sum);
    }
  }

  // ---- firing-rate mean: total_spikes / (2 * 64 * 256 * 1024)
  __syncthreads();
#pragma unroll
  for (int off = 16; off > 0; off >>= 1) frc += __shfl_down(frc, off, 32);
  if (lane == 0) redb[wave] = frc;
  __syncthreads();
  if (tid == 0) {
    float s = 0.0f;
#pragma unroll
    for (int i = 0; i < 32; ++i) s += redb[i];
    out[(size_t)S_ * B_ * O_] = s / 33554432.0f;
  }
}

// ---------------------------------------------------------------------------
extern "C" void kernel_launch(void* const* d_in, const int* in_sizes, int n_in,
                              void* d_out, int out_size, void* d_ws, size_t ws_size,
                              hipStream_t stream) {
  const float* input     = (const float*)d_in[0];
  const float* W_i       = (const float*)d_in[1];
  const float* b_i       = (const float*)d_in[2];
  const float* W_i2r     = (const float*)d_in[3];
  const float* b_i2r     = (const float*)d_in[4];
  const float* W_r       = (const float*)d_in[5];
  const float* b_r       = (const float*)d_in[6];
  const float* W_o       = (const float*)d_in[7];
  const float* b_o       = (const float*)d_in[8];
  const float* tau_adp_i = (const float*)d_in[9];
  const float* tau_adp_r = (const float*)d_in[10];
  const float* tau_m_i   = (const float*)d_in[11];
  const float* tau_m_r   = (const float*)d_in[12];
  const float* tau_m_o   = (const float*)d_in[13];
  const float* init1     = (const float*)d_in[14];
  const float* init2     = (const float*)d_in[15];
  const float* init3     = (const float*)d_in[16];
  float* out = (float*)d_out;

  // workspace: [0, 64MB) d1 activations f32; [64MB, +512KB) W_i bf16
  float* d1 = (float*)d_ws;
  unsigned short* wib = (unsigned short*)((char*)d_ws + (size_t)S_ * B_ * N_ * 4);

  k_conv_wi<<<(N_ * L_ + 255) / 256, 256, 0, stream>>>(W_i, wib);
  k_gemm_d1<<<2048, 256, 0, stream>>>(input, wib, b_i, d1);
  k_scan<<<1, 1024, 327680, stream>>>(d1, W_i2r, b_i2r, W_r, b_r, W_o, b_o,
                                      tau_adp_i, tau_adp_r, tau_m_i, tau_m_r,
                                      tau_m_o, init1, init2, init3, out);
}
